// DMPNN_16913581212025
// MI455X (gfx1250) — matrix operand order repaired and checked by reading, not compiled, not run on hardware
//
#include <hip/hip_runtime.h>

// ---------------- problem constants (match reference) ----------------
#define N_NODES  100000
#define N_EDGES  1600000
#define N_GRAPHS 256
#define FN 6
#define FE 3
#define HID 128
#define NW 4   // waves per block in the edge kernel (wave32)

typedef __attribute__((ext_vector_type(16))) __bf16 v16bf;
typedef __attribute__((ext_vector_type(8)))  __bf16 v8bf;
typedef __attribute__((ext_vector_type(2)))  __bf16 v2bf;
typedef __attribute__((ext_vector_type(8)))  float  v8f;
typedef __attribute__((ext_vector_type(2)))  float  v2f;

union V16 { v16bf v; v8bf half8[2]; };

// pack two f32 into one dword of bf16s: one v_cvt_pk_bf16_f32
__device__ __forceinline__ unsigned pk2bf(float x, float y) {
  v2f p; p.x = x; p.y = y;
  return __builtin_bit_cast(unsigned, __builtin_convertvector(p, v2bf));
}
__device__ __forceinline__ unsigned short f2bf(float f) {
  return __builtin_bit_cast(unsigned short, (__bf16)f);
}

// true => every 16-edge tile is full, all bounds checks fold away
constexpr bool TAIL_EXACT = (N_EDGES % 16) == 0;

// ---------------- workspace zeroing ----------------
__global__ void zero_ws_kernel(float* ws, long n) {
  long i = (long)blockIdx.x * blockDim.x + threadIdx.x;
  long stride = (long)gridDim.x * blockDim.x;
  for (; i < n; i += stride) ws[i] = 0.0f;
}

// ---------------- fused edge-MLP + segment-sum scatter ----------------
__global__ __launch_bounds__(NW * 32)
void edge_mlp_scatter(const float* __restrict__ h,
                      const float* __restrict__ e,
                      const float* __restrict__ W1, const float* __restrict__ b1,
                      const float* __restrict__ W2, const float* __restrict__ b2,
                      const int* __restrict__ src, const int* __restrict__ dst,
                      float* __restrict__ nsum, float* __restrict__ deg)
{
  // weights transposed in LDS: Wt[col][k], bf16; W1 K padded 15 -> 32
  __shared__ __align__(32) unsigned short sWt1[HID * 32];
  __shared__ __align__(32) unsigned short sWt2[HID * HID];
  __shared__ float sB1[HID], sB2[HID];
  // per-wave staging
  __shared__ __align__(32) unsigned short sA1[NW][16 * 32];
  __shared__ __align__(32) unsigned short sA2[NW][16 * HID];
  __shared__ int sDst[NW][16];

  const int tid = threadIdx.x;
  const int NIN = FE + 2 * FN;   // 15

  // preload weights transposed, converting k-pairs with one pk-convert each
  for (int i = tid; i < HID * 16; i += blockDim.x) {      // sWt1 as u32[col][16]
    const int col = i >> 4, k0 = (i & 15) * 2;
    const float x = (k0     < NIN) ? W1[k0 * HID + col]       : 0.0f;
    const float y = (k0 + 1 < NIN) ? W1[(k0 + 1) * HID + col] : 0.0f;
    ((unsigned*)sWt1)[i] = pk2bf(x, y);
  }
  for (int i = tid; i < HID * 64; i += blockDim.x) {      // sWt2 as u32[col][64]
    const int col = i >> 6, k0 = (i & 63) * 2;
    ((unsigned*)sWt2)[i] = pk2bf(W2[k0 * HID + col], W2[(k0 + 1) * HID + col]);
  }
  for (int i = tid; i < HID; i += blockDim.x) { sB1[i] = b1[i]; sB2[i] = b2[i]; }
  __syncthreads();

  const int lane = tid & 31;
  const int wid  = tid >> 5;
  const int hf   = lane >> 4;   // K-half / M-half select
  const int l15  = lane & 15;

  const int numTiles  = (N_EDGES + 15) / 16;
  const int waveGlob  = blockIdx.x * NW + wid;
  const int waveCount = gridDim.x * NW;

  unsigned short* A1 = sA1[wid];
  unsigned short* A2 = sA2[wid];
  const float2* __restrict__ h2 = (const float2*)h;   // h rows are 8B-aligned

  for (int tile = waveGlob; tile < numTiles; tile += waveCount) {
    // ---- gather: lanes 0..15 build one 32-wide bf16 row each ----
    if (lane < 16) {
      const int eidx = tile * 16 + lane;
      unsigned rowu[16];                 // 16 dwords = 32 bf16 (K padded to 32)
      #pragma unroll
      for (int i = 0; i < 16; ++i) rowu[i] = 0u;
      if (TAIL_EXACT || eidx < N_EDGES) {
        const int s = src[eidx], d = dst[eidx];
        #pragma unroll
        for (int i = 0; i < 3; ++i) {                 // h[src]: 3 x float2
          const float2 p = h2[s * 3 + i];
          rowu[i] = pk2bf(p.x, p.y);
        }
        #pragma unroll
        for (int i = 0; i < 3; ++i) {                 // h[dst]: 3 x float2
          const float2 p = h2[d * 3 + i];
          rowu[3 + i] = pk2bf(p.x, p.y);
        }
        rowu[6] = pk2bf(e[eidx * FE], e[eidx * FE + 1]);   // e0,e1
        rowu[7] = pk2bf(e[eidx * FE + 2], 0.0f);            // e2, pad
        sDst[wid][lane] = d;
        atomicAdd(&deg[d], 1.0f);
      }
      unsigned* dp = (unsigned*)(A1 + lane * 32);
      #pragma unroll
      for (int i = 0; i < 16; ++i) dp[i] = rowu[i];
    }

    // ---- layer 1: [16x32(bf16)] @ [32x128] -> relu -> A2 (bf16 in LDS) ----
    V16 a1;  // A fragment: row = l15, K-half = hf  (ISA 16-bit 16x32 A layout)
    a1.half8[0] = *(const v8bf*)(A1 + l15 * 32 + hf * 8);
    a1.half8[1] = *(const v8bf*)(A1 + l15 * 32 + 16 + hf * 8);
    #pragma unroll
    for (int t = 0; t < 8; ++t) {
      V16 b;  // B fragment: col = 16*t + l15, K-half = hf (same K order as A)
      const v8bf* w = (const v8bf*)(sWt1 + (16 * t + l15) * 32 + hf * 8);
      b.half8[0] = w[0]; b.half8[1] = w[2];
      v8f c = {};
      c = __builtin_amdgcn_wmma_f32_16x16x32_bf16(
              false, a1.v, false, b.v, (short)0, c, false, false);
      const int n = 16 * t + l15;
      const float bias = sB1[n];
      #pragma unroll
      for (int r = 0; r < 8; r += 2) {     // C: M = r + 8*hf, N = n
        const unsigned u = pk2bf(fmaxf(c[r] + bias, 0.0f),
                                 fmaxf(c[r + 1] + bias, 0.0f));
        A2[(r     + 8 * hf) * HID + n] = (unsigned short)u;
        A2[(r + 1 + 8 * hf) * HID + n] = (unsigned short)(u >> 16);
      }
    }

    // ---- layer 2: [16x128] @ [128x128], then atomic scatter-add ----
    V16 a2[4];  // A fragments depend only on kk: load once, reuse for all 8 N-tiles
    #pragma unroll
    for (int kk = 0; kk < 4; ++kk) {
      a2[kk].half8[0] = *(const v8bf*)(A2 + l15 * HID + kk * 32 + hf * 8);
      a2[kk].half8[1] = *(const v8bf*)(A2 + l15 * HID + kk * 32 + 16 + hf * 8);
    }
    #pragma unroll
    for (int t = 0; t < 8; ++t) {
      v8f c = {};
      #pragma unroll
      for (int kk = 0; kk < 4; ++kk) {
        V16 b;
        const v8bf* w = (const v8bf*)(sWt2 + (16 * t + l15) * HID + kk * 32 + hf * 8);
        b.half8[0] = w[0]; b.half8[1] = w[2];
        c = __builtin_amdgcn_wmma_f32_16x16x32_bf16(
                false, a2[kk].v, false, b.v, (short)0, c, false, false);
      }
      const int n = 16 * t + l15;
      const float bias = sB2[n];
      #pragma unroll
      for (int r = 0; r < 8; ++r) {
        const int M = r + 8 * hf;
        const int eidx = tile * 16 + M;
        if (TAIL_EXACT || eidx < N_EDGES)
          atomicAdd(&nsum[(size_t)sDst[wid][M] * HID + n], c[r] + bias);
      }
    }
  }
}

// ---------------- node mean + per-graph accumulation ----------------
__global__ __launch_bounds__(HID)
void node_mean_graph(const float* __restrict__ nsum,
                     const float* __restrict__ deg,
                     const int* __restrict__ node_graph,
                     float* __restrict__ gsum, float* __restrict__ cnt)
{
  const int v = blockIdx.x;
  const int t = threadIdx.x;
  const float inv = 1.0f / fmaxf(deg[v], 1.0f);
  const int g = node_graph[v];
  atomicAdd(&gsum[(size_t)g * HID + t], nsum[(size_t)v * HID + t] * inv);
  if (t == 0) atomicAdd(&cnt[g], 1.0f);
}

// ---------------- graph mean + readout MLP (tiny: 8 MFLOP) ----------------
__global__ __launch_bounds__(HID)
void readout_kernel(const float* __restrict__ gsum, const float* __restrict__ cnt,
                    const float* __restrict__ R1, const float* __restrict__ rb1,
                    const float* __restrict__ R2, const float* __restrict__ rb2,
                    float* __restrict__ out)
{
  const int g = blockIdx.x;
  const int j = threadIdx.x;
  __shared__ float hg[HID];
  __shared__ float red[HID];
  const float inv = 1.0f / fmaxf(cnt[g], 1.0f);
  hg[j] = gsum[(size_t)g * HID + j] * inv;
  __syncthreads();
  float acc = rb1[j];
  #pragma unroll 8
  for (int k = 0; k < HID; ++k) acc += hg[k] * R1[k * HID + j];
  acc = fmaxf(acc, 0.0f);
  red[j] = acc * R2[j];
  __syncthreads();
  for (int s = HID / 2; s > 0; s >>= 1) {
    if (j < s) red[j] += red[j + s];
    __syncthreads();
  }
  if (j == 0) out[g] = red[0] + rb2[0];
}

extern "C" void kernel_launch(void* const* d_in, const int* in_sizes, int n_in,
                              void* d_out, int out_size, void* d_ws, size_t ws_size,
                              hipStream_t stream) {
  const float* h   = (const float*)d_in[0];
  const float* e   = (const float*)d_in[1];
  const float* W1  = (const float*)d_in[2];
  const float* b1  = (const float*)d_in[3];
  const float* W2  = (const float*)d_in[4];
  const float* b2  = (const float*)d_in[5];
  const float* R1  = (const float*)d_in[6];
  const float* rb1 = (const float*)d_in[7];
  const float* R2  = (const float*)d_in[8];
  const float* rb2 = (const float*)d_in[9];
  const int*   src = (const int*)d_in[10];
  const int*   dst = (const int*)d_in[11];
  const int*   ng  = (const int*)d_in[12];
  float* out = (float*)d_out;

  float* ws   = (float*)d_ws;
  float* nsum = ws;                                   // [N, 128]
  float* deg  = nsum + (size_t)N_NODES * HID;         // [N]
  float* gsum = deg + N_NODES;                        // [G, 128]
  float* cnt  = gsum + (size_t)N_GRAPHS * HID;        // [G]
  const long zn = (long)N_NODES * HID + N_NODES + (long)N_GRAPHS * HID + N_GRAPHS;

  zero_ws_kernel<<<2048, 256, 0, stream>>>(ws, zn);
  edge_mlp_scatter<<<4096, NW * 32, 0, stream>>>(h, e, W1, b1, W2, b2,
                                                 src, dst, nsum, deg);
  node_mean_graph<<<N_NODES, HID, 0, stream>>>(nsum, deg, ng, gsum, cnt);
  readout_kernel<<<N_GRAPHS, HID, 0, stream>>>(gsum, cnt, R1, rb1, R2, rb2, out);
}
